// GraphAttention_81784767250896
// MI455X (gfx1250) — matrix operand order repaired and checked by reading, not compiled, not run on hardware
//
#include <hip/hip_runtime.h>

// CDNA5 / gfx1250, wave32. Dense GAT attention via v_wmma_f32_16x16x32_f16.

typedef __attribute__((ext_vector_type(16))) _Float16 v16h;
typedef __attribute__((ext_vector_type(8)))  float    v8f;

#define N_NODES 4096
#define F_IN    128
#define UNITS   64
#define HEADS   4
#define JSPLIT  4
#define JRANGE  (N_NODES / JSPLIT)   // 1024 j's per wave
#define NCHUNK  (JRANGE / 32)        // 32 K-chunks of 32

// ---------------- kernel 1: x = inputs @ w  (f32 + f16-transposed copy) ----
__global__ __launch_bounds__(256) void xw_kernel(const float* __restrict__ in,
                                                 const float* __restrict__ w,
                                                 float* __restrict__ x,
                                                 _Float16* __restrict__ xT) {
  int tid = blockIdx.x * 256 + threadIdx.x;        // 4096*64 threads
  int i = tid >> 6, u = tid & 63;
  const float* row = in + (size_t)i * F_IN;
  float acc = 0.f;
#pragma unroll 8
  for (int k = 0; k < F_IN; ++k) acc = fmaf(row[k], w[k * UNITS + u], acc);
  x[(size_t)i * UNITS + u]    = acc;               // f32, for e1/e2
  xT[(size_t)u * N_NODES + i] = (_Float16)acc;     // f16 [U][N] -> WMMA B tiles
}

// -------- kernel 2: e1 = x@aw1 ([N][H]), e2T = (x@aw2)^T ([H][N]) ----------
// e2 is stored transposed so the attention kernel can fetch it with the same
// two contiguous 8-j runs per lane-half that the adj loads use (float4 loads).
__global__ __launch_bounds__(256) void e_kernel(const float* __restrict__ x,
                                                const float* __restrict__ w1,
                                                const float* __restrict__ w2,
                                                float* __restrict__ e1,
                                                float* __restrict__ e2T) {
  int tid = blockIdx.x * 256 + threadIdx.x;        // 4096*4 threads
  int i = tid >> 2, h = tid & 3;
  const float* xr = x + (size_t)i * UNITS;
  float a1 = 0.f, a2 = 0.f;
#pragma unroll 8
  for (int u = 0; u < UNITS; ++u) {
    float xv = xr[u];
    a1 = fmaf(xv, w1[u * HEADS + h], a1);
    a2 = fmaf(xv, w2[u * HEADS + h], a2);
  }
  e1[tid] = a1;                                    // [N][H]
  e2T[(size_t)h * N_NODES + i] = a2;               // [H][N]
}

// ---------------- kernel 3: fused exp-score + WMMA aggregation -------------
// Wave = (16-row i-block, head-pair, j-quarter). Per 32-j chunk:
//   build A[16x32] = 2^(max(adj*(e1+e2),0)*log2e - 8) in f16 (scale cancels),
//   4 WMMAs vs x-tiles (U=64) + 1 WMMA vs ones (row-sum Z), per head.
__global__ __launch_bounds__(256) void attn_kernel(const float* __restrict__ adj,
                                                   const float* __restrict__ e1,
                                                   const float* __restrict__ e2T,
                                                   const _Float16* __restrict__ xT,
                                                   float* __restrict__ accp,   // [JSPLIT][N][H][U]
                                                   float* __restrict__ zp) {   // [JSPLIT][N][H]
  const int lane = threadIdx.x & 31;
  const int gw   = blockIdx.x * 8 + (threadIdx.x >> 5);   // 2048 waves
  const int ib   = gw & 255;
  const int hp   = (gw >> 8) & 1;
  const int js   = gw >> 9;
  const int i0   = ib * 16;
  const int h0   = hp * 2;
  const int m    = lane & 15;       // row within A / column within B,C,D
  const int hi   = lane >> 4;       // lane-half: selects K group / C row group

  const float2 e1v = *(const float2*)(e1 + (i0 + m) * HEADS + h0);

  v16h bones;
#pragma unroll
  for (int t = 0; t < 16; ++t) bones[t] = (_Float16)1.0f;

  v8f c[2][4] = {{{}, {}, {}, {}}, {{}, {}, {}, {}}};
  v8f zc[2]   = {{}, {}};

  const float* adjrow = adj + (size_t)(i0 + m) * N_NODES;
  const float* e2r0   = e2T + (size_t)(h0 + 0) * N_NODES;
  const float* e2r1   = e2T + (size_t)(h0 + 1) * N_NODES;

  for (int jc = 0; jc < NCHUNK; ++jc) {
    const int j0 = js * JRANGE + jc * 32;
    const int cb = j0 + hi * 8;     // base column for this lane-half

    // adj for this lane's 16 A slots: cols cb + {0..7, 16..23}
    const float4* ap  = (const float4*)(adjrow + cb);
    const float4* ap2 = (const float4*)(adjrow + cb + 16);
    float4 a0 = ap[0], a1 = ap[1], a2 = ap2[0], a3 = ap2[1];
    float av[16] = {a0.x, a0.y, a0.z, a0.w, a1.x, a1.y, a1.z, a1.w,
                    a2.x, a2.y, a2.z, a2.w, a3.x, a3.y, a3.z, a3.w};

    // e2 for both heads, same column pattern as adj (broadcast across lanes)
    const float4* p0  = (const float4*)(e2r0 + cb);
    const float4* p0b = (const float4*)(e2r0 + cb + 16);
    const float4* p1  = (const float4*)(e2r1 + cb);
    const float4* p1b = (const float4*)(e2r1 + cb + 16);
    float4 q0 = p0[0], q1 = p0[1], q2 = p0b[0], q3 = p0b[1];
    float4 r0 = p1[0], r1 = p1[1], r2 = p1b[0], r3 = p1b[1];
    float ev0[16] = {q0.x, q0.y, q0.z, q0.w, q1.x, q1.y, q1.z, q1.w,
                     q2.x, q2.y, q2.z, q2.w, q3.x, q3.y, q3.z, q3.w};
    float ev1[16] = {r0.x, r0.y, r0.z, r0.w, r1.x, r1.y, r1.z, r1.w,
                     r2.x, r2.y, r2.z, r2.w, r3.x, r3.y, r3.z, r3.w};

    v16h a[2];
#pragma unroll
    for (int idx = 0; idx < 16; ++idx) {
      float ad = av[idx];
      // score = adj*relu(t) = max(adj*t,0);  E' = 2^(score*log2e - 8)
      float s0 = fminf(fmaxf(ad * (e1v.x + ev0[idx]), 0.0f), 16.0f);
      float s1 = fminf(fmaxf(ad * (e1v.y + ev1[idx]), 0.0f), 16.0f);
      float E0 = __builtin_amdgcn_exp2f(fmaf(s0, 1.44269504089f, -8.0f));
      float E1 = __builtin_amdgcn_exp2f(fmaf(s1, 1.44269504089f, -8.0f));
      a[0][idx] = (_Float16)E0;
      a[1][idx] = (_Float16)E1;
    }

    // B tiles: 32x16, K = hi*8 + {0..7, 16..23}, N = m  -> two 8-f16 runs per lane
    v16h b[4];
#pragma unroll
    for (int ut = 0; ut < 4; ++ut) {
      const _Float16* bp = xT + (size_t)(ut * 16 + m) * N_NODES + j0 + hi * 8;
#pragma unroll
      for (int s = 0; s < 8; ++s) { b[ut][s] = bp[s]; b[ut][s + 8] = bp[16 + s]; }
    }

#pragma unroll
    for (int hh = 0; hh < 2; ++hh) {
#pragma unroll
      for (int ut = 0; ut < 4; ++ut)
        c[hh][ut] = __builtin_amdgcn_wmma_f32_16x16x32_f16(
            false, a[hh], false, b[ut], (short)0, c[hh][ut], false, false);
      zc[hh] = __builtin_amdgcn_wmma_f32_16x16x32_f16(
          false, a[hh], false, bones, (short)0, zc[hh], false, false);
    }
  }

  // Writeback partials (each element has exactly one writer -> deterministic)
  float* accs = accp + (size_t)js * N_NODES * HEADS * UNITS;
  float* zs   = zp + (size_t)js * N_NODES * HEADS;
#pragma unroll
  for (int hh = 0; hh < 2; ++hh) {
    const int h = h0 + hh;
#pragma unroll
    for (int ut = 0; ut < 4; ++ut) {
#pragma unroll
      for (int r = 0; r < 8; ++r) {
        int i = i0 + r + 8 * hi;               // C/D layout: M = r + 8*(lane/16)
        int u = ut * 16 + m;                    // N = lane%16
        accs[((size_t)i * HEADS + h) * UNITS + u] = c[hh][ut][r];
      }
    }
    if (m == 0) {                               // all D columns equal Z
#pragma unroll
      for (int r = 0; r < 8; ++r) {
        int i = i0 + r + 8 * hi;
        zs[(size_t)i * HEADS + h] = zc[hh][r];
      }
    }
  }
}

// ---------------- kernel 4: reduce j-partials, softmax-normalize, relu ----
__global__ __launch_bounds__(256) void final_kernel(const float* __restrict__ accp,
                                                    const float* __restrict__ zp,
                                                    float* __restrict__ out) {
  int tid = blockIdx.x * 256 + threadIdx.x;       // 4096*256
  int i = tid >> 8;
  int h = (tid >> 6) & 3;
  float a = 0.f, z = 0.f;
#pragma unroll
  for (int js = 0; js < JSPLIT; ++js) {
    a += accp[(size_t)js * (N_NODES * HEADS * UNITS) + tid];
    z += zp[(size_t)js * (N_NODES * HEADS) + i * HEADS + h];
  }
  out[tid] = fmaxf(a / z, 0.0f);                  // [N, H*U] head-major
}

// ---------------- host launcher -------------------------------------------
extern "C" void kernel_launch(void* const* d_in, const int* in_sizes, int n_in,
                              void* d_out, int out_size, void* d_ws, size_t ws_size,
                              hipStream_t stream) {
  (void)in_sizes; (void)n_in; (void)out_size; (void)ws_size;
  const float* inputs = (const float*)d_in[0];   // [4096,128]
  const float* adj    = (const float*)d_in[1];   // [4096,4096]
  const float* w      = (const float*)d_in[2];   // [128,64]
  const float* aw1    = (const float*)d_in[3];   // [64,4]
  const float* aw2    = (const float*)d_in[4];   // [64,4]
  float* out = (float*)d_out;                    // [4096,256]

  char* ws = (char*)d_ws;
  float*    xbuf = (float*)(ws + 0);                 //  1,048,576 B
  _Float16* xT   = (_Float16*)(ws + 1048576);        //    524,288 B
  float*    e1   = (float*)(ws + 1572864);           //     65,536 B
  float*    e2T  = (float*)(ws + 1638400);           //     65,536 B
  float*    accp = (float*)(ws + 1703936);           // 16,777,216 B
  float*    zp   = (float*)(ws + 18481152);          //    262,144 B

  xw_kernel<<<(N_NODES * UNITS) / 256, 256, 0, stream>>>(inputs, w, xbuf, xT);
  e_kernel<<<(N_NODES * HEADS) / 256, 256, 0, stream>>>(xbuf, aw1, aw2, e1, e2T);
  attn_kernel<<<256, 256, 0, stream>>>(adj, e1, e2T, xT, accp, zp);
  final_kernel<<<(N_NODES * HEADS * UNITS) / 256, 256, 0, stream>>>(accp, zp, out);
}
